// TransformerBlock_57105885168099
// MI455X (gfx1250) — matrix operand order repaired and checked by reading, not compiled, not run on hardware
//
#include <hip/hip_runtime.h>

// ---------------------------------------------------------------------------
// CDNA5 (gfx1250) transformer block: bf16 WMMA + fp32 accum, double-buffered
// async global->LDS DMA staging, pre-transposed bf16 weights.
// ---------------------------------------------------------------------------

typedef __bf16 bf16;
typedef __attribute__((ext_vector_type(16))) __bf16 v16bf;
typedef __attribute__((ext_vector_type(8)))  __bf16 v8bf;
typedef __attribute__((ext_vector_type(4)))  __bf16 v4bf;
typedef __attribute__((ext_vector_type(8)))  float  v8f;
typedef __attribute__((ext_vector_type(4)))  float  v4f;
typedef int v4i __attribute__((vector_size(16)));

#define WMMA_BF16(a, b, c) \
  __builtin_amdgcn_wmma_f32_16x16x32_bf16(false, (a), false, (b), (short)0, (c), false, false)

#if defined(__has_builtin)
#if __has_builtin(__builtin_amdgcn_global_load_async_to_lds_b128)
#define HAVE_ASYNC_LDS 1
#endif
#endif

#define AS_GLOBAL __attribute__((address_space(1)))
#define AS_SHARED __attribute__((address_space(3)))

// 16-byte global -> LDS copy. Async DMA (ASYNCcnt) when the toolchain has it.
static __device__ inline void cp16_g2l(void* lds_dst, const void* gsrc) {
#ifdef HAVE_ASYNC_LDS
  __builtin_amdgcn_global_load_async_to_lds_b128(
      (AS_GLOBAL v4i*)gsrc, (AS_SHARED v4i*)lds_dst, 0, 0);
#else
  *(v8bf*)lds_dst = *(const v8bf*)gsrc;
#endif
}

template <int N>
static __device__ inline void async_wait_le() {
#ifdef HAVE_ASYNC_LDS
#if __has_builtin(__builtin_amdgcn_s_wait_asynccnt)
  __builtin_amdgcn_s_wait_asynccnt(N);
#else
  asm volatile("s_wait_asynccnt %0" ::"i"(N) : "memory");
#endif
#endif
}

static constexpr int kB  = 4;
static constexpr int kS  = 1024;
static constexpr int kD  = 1024;
static constexpr int kH  = 16;
static constexpr int kHD = 64;
static constexpr int kM  = 2736;
static constexpr int kMp = 2752;     // kM padded to a multiple of 64/32
static constexpr int kT  = kB * kS;  // 4096 tokens
static constexpr float kEPS = 1e-6f;

// ---------------------------------------------------------------------------
// Fragment loaders (wave32 layouts per CDNA5 ISA 7.12.2), 128-bit LDS reads.
// ---------------------------------------------------------------------------

// A: 16x32 bf16 row-major at p[m*ld + k]. Per lane: two contiguous 8-half runs.
static __device__ inline v16bf load_a_frag(const bf16* p, int ld) {
  const int lane = threadIdx.x & 31;
  const int m  = lane & 15;
  const int kb = (lane >> 4) * 8;
  const bf16* q = p + m * ld + kb;
  const v8bf lo = *(const v8bf*)q;         // K = kb .. kb+7
  const v8bf hi = *(const v8bf*)(q + 16);  // K = kb+16 .. kb+23
  return __builtin_shufflevector(lo, hi, 0, 1, 2, 3, 4, 5, 6, 7,
                                 8, 9, 10, 11, 12, 13, 14, 15);
}

// B: 32x16, element (k,n) stored n-major at p[n*ld + k]. Per lane: two contiguous 8-half runs.
static __device__ inline v16bf load_b_frag_t(const bf16* p, int ld) {
  const int lane = threadIdx.x & 31;
  const int n  = lane & 15;
  const int kb = (lane >> 4) * 8;
  const bf16* q = p + n * ld + kb;
  const v8bf lo = *(const v8bf*)q;
  const v8bf hi = *(const v8bf*)(q + 16);
  return __builtin_shufflevector(lo, hi, 0, 1, 2, 3, 4, 5, 6, 7,
                                 8, 9, 10, 11, 12, 13, 14, 15);
}

// ---------------------------------------------------------------------------
// Spectral sigma via power iteration (one workgroup per weight, LDS resident)
// ---------------------------------------------------------------------------

static __device__ void block_normalize(float* vec, int n, float* red) {
  float ss = 0.f;
  for (int i = threadIdx.x; i < n; i += 256) ss += vec[i] * vec[i];
  red[threadIdx.x] = ss;
  __syncthreads();
  for (int s = 128; s > 0; s >>= 1) {
    if ((int)threadIdx.x < s) red[threadIdx.x] += red[threadIdx.x + s];
    __syncthreads();
  }
  const float inv = 1.f / (sqrtf(red[0]) + kEPS);
  for (int i = threadIdx.x; i < n; i += 256) vec[i] *= inv;
  __syncthreads();
}

__global__ __launch_bounds__(256)
void sigma_all_kernel(const float* w0, const float* w1, const float* w2,
                      const float* w3, const float* w4, const float* w5,
                      const float* w6,
                      const float* sp0, const float* sp1, const float* sp2,
                      const float* sp3, const float* sp4, const float* sp5,
                      const float* sp6,
                      float* scales) {
  __shared__ float u[2752];
  __shared__ float v[2752];
  __shared__ float red[256];

  const float* W; const float* sp; int R, C;
  switch (blockIdx.x) {
    case 0: W = w0; sp = sp0; R = kD; C = kD; break;   // wq
    case 1: W = w1; sp = sp1; R = kD; C = kD; break;   // wk
    case 2: W = w2; sp = sp2; R = kD; C = kD; break;   // wv
    case 3: W = w3; sp = sp3; R = kD; C = kD; break;   // wo
    case 4: W = w4; sp = sp4; R = kD; C = kM; break;   // wg
    case 5: W = w5; sp = sp5; R = kD; C = kM; break;   // wu
    default: W = w6; sp = sp6; R = kM; C = kD; break;  // wd
  }

  for (int c = threadIdx.x; c < C; c += 256) u[c] = 1.0f;
  __syncthreads();
  block_normalize(u, C, red);

  for (int it = 0; it < 5; ++it) {
    for (int r = threadIdx.x; r < R; r += 256) {
      float s = 0.f;
      for (int c = 0; c < C; c += 4) {
        const v4f wv = *(const v4f*)&W[(size_t)r * C + c];
        s += wv[0] * u[c] + wv[1] * u[c + 1] + wv[2] * u[c + 2] + wv[3] * u[c + 3];
      }
      v[r] = s;
    }
    __syncthreads();
    block_normalize(v, R, red);
    for (int c = threadIdx.x; c < C; c += 256) {
      float s = 0.f;
      for (int r = 0; r < R; ++r) s += W[(size_t)r * C + c] * v[r];
      u[c] = s;
    }
    __syncthreads();
    block_normalize(u, C, red);
  }

  for (int r = threadIdx.x; r < R; r += 256) {
    float s = 0.f;
    for (int c = 0; c < C; c += 4) {
      const v4f wv = *(const v4f*)&W[(size_t)r * C + c];
      s += wv[0] * u[c] + wv[1] * u[c + 1] + wv[2] * u[c + 2] + wv[3] * u[c + 3];
    }
    v[r] = s;
  }
  __syncthreads();
  float ss = 0.f;
  for (int r = threadIdx.x; r < R; r += 256) ss += v[r] * v[r];
  red[threadIdx.x] = ss;
  __syncthreads();
  for (int s = 128; s > 0; s >>= 1) {
    if ((int)threadIdx.x < s) red[threadIdx.x] += red[threadIdx.x + s];
    __syncthreads();
  }
  if (threadIdx.x == 0) {
    const float nrm = sqrtf(red[0]);
    float sigma = red[0] / (nrm + kEPS);       // v_norm . (W u)
    sigma = fmaxf(sigma, kEPS);
    scales[blockIdx.x] = sp[0] / sigma;
  }
}

// ---------------------------------------------------------------------------
// WT[Npad][Kpad] (bf16) = scale * W[K][N]^T, zero pad beyond K/N.
// Grid: (Kpad/64, Npad/64), 256 threads, LDS-tiled transpose.
// ---------------------------------------------------------------------------

__global__ __launch_bounds__(256)
void scale_transpose_bf16_kernel(const float* __restrict__ W,
                                 const float* __restrict__ scale,
                                 bf16* __restrict__ WT, int K, int N, int Kpad) {
  __shared__ __align__(16) bf16 Ts[64][72];
  const float s = scale[0];
  const int r0 = blockIdx.x * 64;  // k
  const int c0 = blockIdx.y * 64;  // n
  const int tid = threadIdx.x;

  const int lr = tid >> 4;        // 0..15
  const int lc = (tid & 15) * 4;  // 0..60
#pragma unroll
  for (int e = 0; e < 4; ++e) {
    const int r = lr + e * 16;
    const int gr = r0 + r, gc = c0 + lc;
    v4f x = {};
    if (gr < K && gc + 4 <= N) x = *(const v4f*)&W[(size_t)gr * N + gc];
    Ts[r][lc]     = (bf16)(x[0] * s);
    Ts[r][lc + 1] = (bf16)(x[1] * s);
    Ts[r][lc + 2] = (bf16)(x[2] * s);
    Ts[r][lc + 3] = (bf16)(x[3] * s);
  }
  __syncthreads();

  const int oc = (tid & 7) * 8;  // k-offset within tile
  const int orow = tid >> 3;     // 0..31 (+32)
#pragma unroll
  for (int e = 0; e < 2; ++e) {
    const int c = orow + e * 32;
    v8bf o;
#pragma unroll
    for (int j = 0; j < 8; ++j) o[j] = Ts[oc + j][c];
    *(v8bf*)&WT[(size_t)(c0 + c) * Kpad + r0 + oc] = o;
  }
}

// ---------------------------------------------------------------------------
// RMSNorm (rows of length D, D % 4 == 0) -> bf16
// ---------------------------------------------------------------------------

__global__ __launch_bounds__(256)
void rmsnorm_rows_kernel(const float* __restrict__ X, const float* __restrict__ scale,
                         bf16* __restrict__ Y, int D) {
  __shared__ float red[256];
  const int row = blockIdx.x;
  const float* p = X + (size_t)row * D;
  float ss = 0.f;
  for (int i = threadIdx.x * 4; i < D; i += 1024) {
    const v4f x = *(const v4f*)&p[i];
    ss += x[0] * x[0] + x[1] * x[1] + x[2] * x[2] + x[3] * x[3];
  }
  red[threadIdx.x] = ss;
  __syncthreads();
  for (int s = 128; s > 0; s >>= 1) {
    if ((int)threadIdx.x < s) red[threadIdx.x] += red[threadIdx.x + s];
    __syncthreads();
  }
  const float r = rsqrtf(red[0] / (float)D + kEPS);
  bf16* q = Y + (size_t)row * D;
  for (int i = threadIdx.x * 4; i < D; i += 1024) {
    const v4f x  = *(const v4f*)&p[i];
    const v4f sc = *(const v4f*)&scale[i];
    v4bf o;
    o[0] = (bf16)(x[0] * r * sc[0]); o[1] = (bf16)(x[1] * r * sc[1]);
    o[2] = (bf16)(x[2] * r * sc[2]); o[3] = (bf16)(x[3] * r * sc[3]);
    *(v4bf*)&q[i] = o;
  }
}

// per-head RMSNorm over HD=64 (one wave32 per segment) -> bf16
__global__ __launch_bounds__(256)
void head_rmsnorm_kernel(const float* __restrict__ X, const float* __restrict__ scale,
                         bf16* __restrict__ Y, int nseg) {
  const int seg  = (blockIdx.x * 256 + threadIdx.x) >> 5;
  const int lane = threadIdx.x & 31;
  if (seg >= nseg) return;
  const float* p = X + (size_t)seg * kHD;
  const float e0 = p[lane], e1 = p[lane + 32];
  float ss = e0 * e0 + e1 * e1;
#pragma unroll
  for (int m = 1; m < 32; m <<= 1) ss += __shfl_xor(ss, m, 32);
  const float r = rsqrtf(ss * (1.f / (float)kHD) + kEPS);
  bf16* q = Y + (size_t)seg * kHD;
  q[lane]      = (bf16)(e0 * r * scale[lane]);
  q[lane + 32] = (bf16)(e1 * r * scale[lane + 32]);
}

// ---------------------------------------------------------------------------
// bf16 WMMA GEMM: C[M,N] = A[M,*] @ BT[N,*]^T (+bias, +residual)
// A: [M][lda] row-major. BT: [>=N][ldb] n-major (pre-transposed weights).
// 256 threads = 8 waves, 64x64 tile, K step 32, double-buffered async DMA.
// ep: 0 = f32 out (+bias), 1 = bf16 out (+bias), 2 = f32 out (+bias +resid),
//     3 = bf16 out (+bias) scattered to [B,H,HD,S] (for attention V).
// Kiter % 32 == 0; all tile addresses in-bounds by construction (padding).
// ---------------------------------------------------------------------------

__global__ __launch_bounds__(256)
void gemm_bf16_kernel(const bf16* __restrict__ A, const bf16* __restrict__ BT,
                      const float* __restrict__ bias, const float* __restrict__ resid,
                      float* __restrict__ outF, bf16* __restrict__ outB,
                      int M, int N, int Kiter, int lda, int ldb, int ldo, int ep) {
  __shared__ __align__(16) bf16 As[2][64][48];
  __shared__ __align__(16) bf16 Bs[2][64][48];
  const int tid  = threadIdx.x;
  const int w    = tid >> 5;
  const int lane = tid & 31;
  const int mrow0 = blockIdx.y * 64;
  const int ncol0 = blockIdx.x * 64;
  const int wm = (w >> 2) * 16;
  const int wn = (w & 3) * 16;
  const int sr = tid >> 2, sc = (tid & 3) * 8;  // one 8-elem vector per tile

  v8f acc0 = {}, acc1 = {};

  auto stage = [&](int kt, int buf) {
    cp16_g2l(&As[buf][sr][sc], &A[(size_t)(mrow0 + sr) * lda + kt + sc]);
    cp16_g2l(&Bs[buf][sr][sc], &BT[(size_t)(ncol0 + sr) * ldb + kt + sc]);
  };

  stage(0, 0);
  const int nIter = Kiter / 32;
  for (int it = 0; it < nIter; ++it) {
    const int cur = it & 1;
    const int kt = it * 32;
    __syncthreads();  // prior readers of buf cur^1 are done
    if (it + 1 < nIter) {
      __builtin_prefetch(&A[(size_t)(mrow0 + sr) * lda + kt + 64 + sc], 0, 3);
      __builtin_prefetch(&BT[(size_t)(ncol0 + sr) * ldb + kt + 64 + sc], 0, 3);
      stage(kt + 32, cur ^ 1);
      async_wait_le<2>();  // this stage's 2 copies still in flight; cur landed
    } else {
      async_wait_le<0>();
    }
    __syncthreads();

    const v16bf bfrag = load_b_frag_t(&Bs[cur][wn][0], 48);
    const v16bf a0 = load_a_frag(&As[cur][wm][0], 48);
    const v16bf a1 = load_a_frag(&As[cur][wm + 32][0], 48);
    acc0 = WMMA_BF16(a0, bfrag, acc0);
    acc1 = WMMA_BF16(a1, bfrag, acc1);
  }

  const int n  = lane & 15;
  const int mb = (lane >> 4) * 8;
  const int col = ncol0 + wn + n;
  if (col < N) {
    const float bv = bias ? bias[col] : 0.f;
#pragma unroll
    for (int t = 0; t < 2; ++t) {
      const v8f& acc = t ? acc1 : acc0;
#pragma unroll
      for (int i = 0; i < 8; ++i) {
        const int row = mrow0 + t * 32 + wm + mb + i;
        const float val = acc[i] + bv;
        if (ep == 3) {
          const int bb = row >> 10, ssq = row & (kS - 1);
          const int hh = col >> 6, hd = col & (kHD - 1);
          outB[(((size_t)bb * kH + hh) * kHD + hd) * kS + ssq] = (bf16)val;
        } else {
          const size_t idx = (size_t)row * ldo + col;
          if (ep == 2)      outF[idx] = val + resid[idx];
          else if (ep == 1) outB[idx] = (bf16)val;
          else              outF[idx] = val;
        }
      }
    }
  }
}

// ---------------------------------------------------------------------------
// Fused gate/up GEMM + SwiGLU: H = silu(A@Wg + bg) * (A@Wu + bu)  -> bf16
// Output row stride ldo (= kMp), pad columns [N, ldo) written as zeros.
// ---------------------------------------------------------------------------

__global__ __launch_bounds__(256)
void gateup_swiglu_kernel(const bf16* __restrict__ A, const bf16* __restrict__ BgT,
                          const bf16* __restrict__ BuT, const float* __restrict__ bg,
                          const float* __restrict__ bu, bf16* __restrict__ Hout,
                          int M, int N, int Kiter, int lda, int ldb, int ldo) {
  __shared__ __align__(16) bf16 As[2][64][48];
  __shared__ __align__(16) bf16 Gs[2][64][48];
  __shared__ __align__(16) bf16 Us[2][64][48];
  const int tid  = threadIdx.x;
  const int w    = tid >> 5;
  const int lane = tid & 31;
  const int mrow0 = blockIdx.y * 64;
  const int ncol0 = blockIdx.x * 64;
  const int wm = (w >> 2) * 16;
  const int wn = (w & 3) * 16;
  const int sr = tid >> 2, sc = (tid & 3) * 8;

  v8f accg0 = {}, accg1 = {}, accu0 = {}, accu1 = {};

  auto stage = [&](int kt, int buf) {
    cp16_g2l(&As[buf][sr][sc], &A[(size_t)(mrow0 + sr) * lda + kt + sc]);
    cp16_g2l(&Gs[buf][sr][sc], &BgT[(size_t)(ncol0 + sr) * ldb + kt + sc]);
    cp16_g2l(&Us[buf][sr][sc], &BuT[(size_t)(ncol0 + sr) * ldb + kt + sc]);
  };

  stage(0, 0);
  const int nIter = Kiter / 32;
  for (int it = 0; it < nIter; ++it) {
    const int cur = it & 1;
    const int kt = it * 32;
    __syncthreads();
    if (it + 1 < nIter) {
      stage(kt + 32, cur ^ 1);
      async_wait_le<3>();
    } else {
      async_wait_le<0>();
    }
    __syncthreads();

    const v16bf bg_f = load_b_frag_t(&Gs[cur][wn][0], 48);
    const v16bf bu_f = load_b_frag_t(&Us[cur][wn][0], 48);
    const v16bf a0 = load_a_frag(&As[cur][wm][0], 48);
    const v16bf a1 = load_a_frag(&As[cur][wm + 32][0], 48);
    accg0 = WMMA_BF16(a0, bg_f, accg0);
    accg1 = WMMA_BF16(a1, bg_f, accg1);
    accu0 = WMMA_BF16(a0, bu_f, accu0);
    accu1 = WMMA_BF16(a1, bu_f, accu1);
  }

  const int n  = lane & 15;
  const int mb = (lane >> 4) * 8;
  const int col = ncol0 + wn + n;
  const bool live = (col < N);
  const float bgv = live ? bg[col] : 0.f;
  const float buv = live ? bu[col] : 0.f;
#pragma unroll
  for (int t = 0; t < 2; ++t) {
    const v8f& ag = t ? accg1 : accg0;
    const v8f& au = t ? accu1 : accu0;
#pragma unroll
    for (int i = 0; i < 8; ++i) {
      const int row = mrow0 + t * 32 + wm + mb + i;
      const float g = ag[i] + bgv;
      const float u = au[i] + buv;
      const float hv = live ? (g / (1.f + __expf(-g))) * u : 0.f;  // zero-fill pad
      Hout[(size_t)row * ldo + col] = (bf16)hv;
    }
  }
}

// ---------------------------------------------------------------------------
// Flash attention with tanh softcap. One block (4 waves) = 64 q-rows of one
// (b,h). K/V streamed in 32-key chunks, double-buffered async DMA.
// Q/K: [B,S,H,HD] bf16.  Vt: [B,H,HD,S] bf16 (pre-transposed by V GEMM).
// O:  [B,S,H,HD] bf16.
// ---------------------------------------------------------------------------

__global__ __launch_bounds__(128)
void attention_kernel(const bf16* __restrict__ Q, const bf16* __restrict__ Kx,
                      const bf16* __restrict__ Vt, bf16* __restrict__ O) {
  __shared__ __align__(16) bf16 Qs[64][72];
  __shared__ __align__(16) bf16 Ks[2][32][72];
  __shared__ __align__(16) bf16 Vs[2][64][48];   // hd x key
  __shared__ __align__(16) bf16 Ps[4][16][48];

  const int bh = blockIdx.y;
  const int b  = bh >> 4;
  const int h  = bh & 15;
  const int q0 = blockIdx.x * 64;
  const int tid  = threadIdx.x;
  const int w    = tid >> 5;
  const int lane = tid & 31;

  auto stageKV = [&](int kt, int buf) {
#pragma unroll
    for (int vv = tid; vv < 256; vv += 128) {
      const int r = vv >> 3, c = (vv & 7) * 8;
      cp16_g2l(&Ks[buf][r][c], &Kx[(((size_t)b * kS + kt + r) * kH + h) * kHD + c]);
    }
#pragma unroll
    for (int vv = tid; vv < 256; vv += 128) {
      const int r = vv >> 2, c = (vv & 3) * 8;   // r: hd, c: key
      cp16_g2l(&Vs[buf][r][c], &Vt[((size_t)bh * kHD + r) * kS + kt + c]);
    }
  };

  // prologue: Q tile (4 vectors/thread) + first K/V chunk (4 vectors/thread)
#pragma unroll
  for (int vv = tid; vv < 512; vv += 128) {
    const int r = vv >> 3, c = (vv & 7) * 8;
    cp16_g2l(&Qs[r][c], &Q[(((size_t)b * kS + q0 + r) * kH + h) * kHD + c]);
  }
  stageKV(0, 0);

  float m_run[8], l_run[8];
#pragma unroll
  for (int i = 0; i < 8; ++i) { m_run[i] = -1e30f; l_run[i] = 0.f; }
  v8f o0 = {}, o1 = {}, o2 = {}, o3 = {};

  const int nIter = kS / 32;
  for (int it = 0; it < nIter; ++it) {
    const int cur = it & 1;
    __syncthreads();  // prior readers of buf cur^1 done
    if (it + 1 < nIter) {
      stageKV((it + 1) * 32, cur ^ 1);
      async_wait_le<4>();
    } else {
      async_wait_le<0>();
    }
    __syncthreads();

    // S = Q K^T for this wave's 16 q-rows x 32 keys
    v8f s0 = {}, s1 = {};
#pragma unroll
    for (int kk = 0; kk < 64; kk += 32) {
      const v16bf aq = load_a_frag(&Qs[w * 16][kk], 72);
      s0 = WMMA_BF16(aq, load_b_frag_t(&Ks[cur][0][kk],  72), s0);
      s1 = WMMA_BF16(aq, load_b_frag_t(&Ks[cur][16][kk], 72), s1);
    }

    // softcap + online softmax (row reductions across 16-lane halves)
    float p0[8], p1[8];
#pragma unroll
    for (int i = 0; i < 8; ++i) {
      const float x0 = 50.f * tanhf(s0[i] * (0.125f * 0.02f));
      const float x1 = 50.f * tanhf(s1[i] * (0.125f * 0.02f));
      float mx = fmaxf(x0, x1);
#pragma unroll
      for (int msk = 1; msk < 16; msk <<= 1) mx = fmaxf(mx, __shfl_xor(mx, msk, 32));
      const float mn    = fmaxf(m_run[i], mx);
      const float alpha = __expf(m_run[i] - mn);
      m_run[i] = mn;
      p0[i] = __expf(x0 - mn);
      p1[i] = __expf(x1 - mn);
      float sum = p0[i] + p1[i];
#pragma unroll
      for (int msk = 1; msk < 16; msk <<= 1) sum += __shfl_xor(sum, msk, 32);
      l_run[i] = l_run[i] * alpha + sum;
      o0[i] *= alpha; o1[i] *= alpha; o2[i] *= alpha; o3[i] *= alpha;
    }

    // repack P (C layout) -> A-fragment staging in LDS
    {
      const int n  = lane & 15;
      const int mb = (lane >> 4) * 8;
#pragma unroll
      for (int i = 0; i < 8; ++i) {
        Ps[w][mb + i][n]      = (bf16)p0[i];
        Ps[w][mb + i][16 + n] = (bf16)p1[i];
      }
    }
    __syncthreads();

    // O += P @ V  (B-fragments from transposed V: contiguous 128-bit reads)
    const v16bf ap = load_a_frag(&Ps[w][0][0], 48);
    o0 = WMMA_BF16(ap, load_b_frag_t(&Vs[cur][0][0],  48), o0);
    o1 = WMMA_BF16(ap, load_b_frag_t(&Vs[cur][16][0], 48), o1);
    o2 = WMMA_BF16(ap, load_b_frag_t(&Vs[cur][32][0], 48), o2);
    o3 = WMMA_BF16(ap, load_b_frag_t(&Vs[cur][48][0], 48), o3);
  }

  const int n  = lane & 15;
  const int mb = (lane >> 4) * 8;
#pragma unroll
  for (int i = 0; i < 8; ++i) {
    const float inv_l = 1.f / l_run[i];
    const int row = q0 + w * 16 + mb + i;
    const size_t base = (((size_t)b * kS + row) * kH + h) * kHD;
    O[base + n]      = (bf16)(o0[i] * inv_l);
    O[base + 16 + n] = (bf16)(o1[i] * inv_l);
    O[base + 32 + n] = (bf16)(o2[i] * inv_l);
    O[base + 48 + n] = (bf16)(o3[i] * inv_l);
  }
}

// ---------------------------------------------------------------------------
// Host-side orchestration
// ---------------------------------------------------------------------------

extern "C" void kernel_launch(void* const* d_in, const int* in_sizes, int n_in,
                              void* d_out, int out_size, void* d_ws, size_t ws_size,
                              hipStream_t stream) {
  (void)in_sizes; (void)n_in; (void)out_size; (void)ws_size;
  const float* x   = (const float*)d_in[0];
  const float* ln1 = (const float*)d_in[1];
  const float* wq  = (const float*)d_in[2];
  const float* sq  = (const float*)d_in[3];
  const float* bq  = (const float*)d_in[4];
  const float* wk  = (const float*)d_in[5];
  const float* sk  = (const float*)d_in[6];
  const float* bk  = (const float*)d_in[7];
  const float* wv  = (const float*)d_in[8];
  const float* sv  = (const float*)d_in[9];
  const float* bv  = (const float*)d_in[10];
  const float* qns = (const float*)d_in[11];
  const float* kns = (const float*)d_in[12];
  const float* wo  = (const float*)d_in[13];
  const float* so  = (const float*)d_in[14];
  const float* bo  = (const float*)d_in[15];
  const float* ln2 = (const float*)d_in[16];
  const float* wg  = (const float*)d_in[17];
  const float* sg  = (const float*)d_in[18];
  const float* bg  = (const float*)d_in[19];
  const float* wu  = (const float*)d_in[20];
  const float* su  = (const float*)d_in[21];
  const float* bu  = (const float*)d_in[22];
  const float* wd  = (const float*)d_in[23];
  const float* sd  = (const float*)d_in[24];
  const float* bd  = (const float*)d_in[25];
  float* out = (float*)d_out;

  char* p = (char*)d_ws;
  auto take = [&](size_t bytes) -> void* {
    void* r = (void*)p;
    p += (bytes + 255) & ~(size_t)255;
    return r;
  };
  float* scales = (float*)take(7 * sizeof(float));
  bf16* wq_t = (bf16*)take((size_t)kD * kD * 2);    // [N=1024][K=1024]
  bf16* wk_t = (bf16*)take((size_t)kD * kD * 2);
  bf16* wv_t = (bf16*)take((size_t)kD * kD * 2);
  bf16* wo_t = (bf16*)take((size_t)kD * kD * 2);
  bf16* wg_t = (bf16*)take((size_t)kMp * kD * 2);   // [Npad=2752][K=1024]
  bf16* wu_t = (bf16*)take((size_t)kMp * kD * 2);
  bf16* wd_t = (bf16*)take((size_t)kD * kMp * 2);   // [N=1024][Kpad=2752]
  bf16* nrm_b = (bf16*)take((size_t)kT * kD * 2);   // attn_in, later mlp_in
  float* fbuf = (float*)take((size_t)kT * kD * 4);  // q_f32 -> k_f32 -> x1
  bf16* qb  = (bf16*)take((size_t)kT * kD * 2);
  bf16* kb  = (bf16*)take((size_t)kT * kD * 2);
  bf16* vbt = (bf16*)take((size_t)kT * kD * 2);     // [B,H,HD,S]
  bf16* ob  = (bf16*)take((size_t)kT * kD * 2);
  bf16* hb  = (bf16*)take((size_t)kT * kMp * 2);    // [4096][2752], pad zeroed

  sigma_all_kernel<<<7, 256, 0, stream>>>(wq, wk, wv, wo, wg, wu, wd,
                                          sq, sk, sv, so, sg, su, sd, scales);

  const dim3 gSq(kD / 64, kD / 64);  // (16,16)
  scale_transpose_bf16_kernel<<<gSq, 256, 0, stream>>>(wq, scales + 0, wq_t, kD, kD, kD);
  scale_transpose_bf16_kernel<<<gSq, 256, 0, stream>>>(wk, scales + 1, wk_t, kD, kD, kD);
  scale_transpose_bf16_kernel<<<gSq, 256, 0, stream>>>(wv, scales + 2, wv_t, kD, kD, kD);
  scale_transpose_bf16_kernel<<<gSq, 256, 0, stream>>>(wo, scales + 3, wo_t, kD, kD, kD);
  scale_transpose_bf16_kernel<<<dim3(kD / 64, kMp / 64), 256, 0, stream>>>(
      wg, scales + 4, wg_t, kD, kM, kD);
  scale_transpose_bf16_kernel<<<dim3(kD / 64, kMp / 64), 256, 0, stream>>>(
      wu, scales + 5, wu_t, kD, kM, kD);
  scale_transpose_bf16_kernel<<<dim3(kMp / 64, kD / 64), 256, 0, stream>>>(
      wd, scales + 6, wd_t, kM, kD, kMp);

  rmsnorm_rows_kernel<<<kT, 256, 0, stream>>>(x, ln1, nrm_b, kD);

  const dim3 gDD(kD / 64, kT / 64);  // (16, 64)
  gemm_bf16_kernel<<<gDD, 256, 0, stream>>>(nrm_b, wq_t, bq, nullptr, fbuf, nullptr,
                                            kT, kD, kD, kD, kD, kD, 0);
  head_rmsnorm_kernel<<<(kT * kH) / 8, 256, 0, stream>>>(fbuf, qns, qb, kT * kH);
  gemm_bf16_kernel<<<gDD, 256, 0, stream>>>(nrm_b, wk_t, bk, nullptr, fbuf, nullptr,
                                            kT, kD, kD, kD, kD, kD, 0);
  head_rmsnorm_kernel<<<(kT * kH) / 8, 256, 0, stream>>>(fbuf, kns, kb, kT * kH);
  gemm_bf16_kernel<<<gDD, 256, 0, stream>>>(nrm_b, wv_t, bv, nullptr, nullptr, vbt,
                                            kT, kD, kD, kD, kD, kD, 3);

  attention_kernel<<<dim3(kS / 64, kB * kH), 128, 0, stream>>>(qb, kb, vbt, ob);

  gemm_bf16_kernel<<<gDD, 256, 0, stream>>>(ob, wo_t, bo, x, fbuf, nullptr,
                                            kT, kD, kD, kD, kD, kD, 2);

  rmsnorm_rows_kernel<<<kT, 256, 0, stream>>>(fbuf, ln2, nrm_b, kD);

  gateup_swiglu_kernel<<<dim3(kMp / 64, kT / 64), 256, 0, stream>>>(
      nrm_b, wg_t, wu_t, bg, bu, hb, kT, kM, kD, kD, kD, kMp);

  gemm_bf16_kernel<<<gDD, 256, 0, stream>>>(hb, wd_t, bd, fbuf, out, nullptr,
                                            kT, kD, kMp, kMp, kMp, kD, 2);
}
